// Attention_82892868812901
// MI455X (gfx1250) — matrix operand order repaired and checked by reading, not compiled, not run on hardware
//
#include <hip/hip_runtime.h>

// ---- CDNA5 WMMA types ----
typedef __attribute__((ext_vector_type(16))) __bf16 bf16x16;
typedef __attribute__((ext_vector_type(8)))  __bf16 bf16x8;
typedef __attribute__((ext_vector_type(8)))  float  f32x8;
typedef __attribute__((ext_vector_type(8)))  float  f32x8v;

#define WMMA_BF16(a, b, c) \
  __builtin_amdgcn_wmma_f32_16x16x32_bf16(false, (a), false, (b), (short)0, (c), false, false)

// Problem constants
#define BQ   4
#define NQ   2048
#define CQ   1024
#define HQ   16
#define DQ   64

// ---- CDNA5 async global->LDS copy (ASYNCcnt path) ----
// __shared__ generic pointers carry the LDS offset in the low 32 bits
// (SHARED aperture base in the high 32), so truncation gives the LDS addr.
static __device__ __forceinline__ void async_copy_b128(void* lds_dst, const void* gsrc) {
  unsigned l = (unsigned)(unsigned long long)(uintptr_t)lds_dst;
  unsigned long long g = (unsigned long long)(uintptr_t)gsrc;
  asm volatile("global_load_async_to_lds_b128 %0, %1, off"
               :: "v"(l), "v"(g) : "memory");
}
static __device__ __forceinline__ void wait_async0() {
  asm volatile("s_wait_asynccnt 0x0" ::: "memory");
}

// =====================================================================
// GEMM: Y[M,F] = X[M,K] @ W[F,K]^T   (f32 in, bf16 WMMA, f32 accum)
// mode 0: Y f32 (+bias)      mode 1: scatter bf16 into q/k/v [B,H,N,D]
// Block tile: 128(M) x 128(F), BK=32. 256 threads = 8 waves; each wave
// owns a 32x64 region = 2 M-tiles x 4 F-tiles -> 8 WMMAs per K-step.
// =====================================================================
__global__ __launch_bounds__(256) void gemm_xwT_kernel(
    const float* __restrict__ X, const float* __restrict__ W,
    const float* __restrict__ bias, float* __restrict__ Y,
    __bf16* __restrict__ qb, __bf16* __restrict__ kb, __bf16* __restrict__ vb,
    int M, int Fdim, int K, int mode)
{
  constexpr int BM = 128, BF = 128, BK = 32, LDP = 40; // pad: 80B rows, 16B aligned
  __shared__ __bf16 sA[BM * LDP];
  __shared__ __bf16 sB[BF * LDP];

  const int tid  = threadIdx.x;
  const int wave = tid >> 5, lane = tid & 31;
  const int hs = lane >> 4, ln = lane & 15;
  const int m0 = blockIdx.y * BM, f0 = blockIdx.x * BF;
  const int rt  = (wave >> 1) * 2;  // first of 2 row tiles (16 rows each)
  const int fj0 = (wave & 1) * 4;   // first of 4 col tiles

  f32x8 acc[2][4] = {};

  for (int kt = 0; kt < K; kt += BK) {
    __syncthreads();
    // cooperative load+convert: 16 elems per thread per matrix (2x f32x8)
#pragma unroll
    for (int c = 0; c < 2; c++) {
      int e0  = tid * 16 + c * 8;
      int row = e0 >> 5, col = e0 & 31;
      f32x8v fa = *reinterpret_cast<const f32x8v*>(X + (size_t)(m0 + row) * K + kt + col);
      f32x8v fb = *reinterpret_cast<const f32x8v*>(W + (size_t)(f0 + row) * K + kt + col);
      bf16x8 ta, tb;
#pragma unroll
      for (int i = 0; i < 8; i++) { ta[i] = (__bf16)fa[i]; tb[i] = (__bf16)fb[i]; }
      *reinterpret_cast<bf16x8*>(&sA[row * LDP + col]) = ta;
      *reinterpret_cast<bf16x8*>(&sB[row * LDP + col]) = tb;
    }
    __syncthreads();

    // A fragments: 16x32 bf16 (lane = row, ISA element->K mapping)
    bf16x16 af[2];
#pragma unroll
    for (int t = 0; t < 2; t++) {
#pragma unroll
      for (int i = 0; i < 16; i++) {
        int kk = ((i >> 3) << 4) + (hs << 3) + (i & 7);
        af[t][i] = sA[((rt + t) * 16 + ln) * LDP + kk];
      }
    }
    // B fragments: 32x16 bf16 (lane = col, element->K mapping)
#pragma unroll
    for (int fj = 0; fj < 4; fj++) {
      bf16x16 bfr;
#pragma unroll
      for (int i = 0; i < 16; i++) {
        int kk = ((i >> 3) << 4) + (hs << 3) + (i & 7);
        bfr[i] = sB[((fj0 + fj) * 16 + ln) * LDP + kk];
      }
      acc[0][fj] = WMMA_BF16(af[0], bfr, acc[0][fj]);
      acc[1][fj] = WMMA_BF16(af[1], bfr, acc[1][fj]);
    }
  }

  // epilogue: C/D layout -> VGPR r holds row r + hs*8, lane&15 = col
#pragma unroll
  for (int t = 0; t < 2; t++) {
#pragma unroll
    for (int fj = 0; fj < 4; fj++) {
#pragma unroll
      for (int r = 0; r < 8; r++) {
        int mrow = m0 + (rt + t) * 16 + r + hs * 8;
        int f    = f0 + (fj0 + fj) * 16 + ln;
        float val = acc[t][fj][r];
        if (mode == 0) {
          Y[(size_t)mrow * Fdim + f] = val + (bias ? bias[f] : 0.0f);
        } else {
          int s = f >> 10, rem = f & 1023, hh = rem >> 6, dd = rem & 63;
          int bb = mrow >> 11, nn = mrow & 2047;
          __bf16* dst = (s == 0) ? qb : (s == 1) ? kb : vb;
          dst[(((size_t)bb * HQ + hh) * NQ + nn) * DQ + dd] = (__bf16)val;
        }
      }
    }
  }
}

// =====================================================================
// RoPE in-place on bf16 q,k in [B,H,N,D]; each thread owns a (d, d+32)
// pair -> no read/write races.
// =====================================================================
__global__ __launch_bounds__(256) void rope_kernel(
    __bf16* __restrict__ q, __bf16* __restrict__ k,
    const float* __restrict__ cosv, const float* __restrict__ sinv)
{
  int idx = blockIdx.x * 256 + threadIdx.x;  // 2^22 total
  int d  = idx & 31;
  int n  = (idx >> 5) & (NQ - 1);
  int bh = idx >> 16;  // 0..63
  size_t base = ((size_t)bh * NQ + n) * DQ;
  float c1 = cosv[n * DQ + d],      s1 = sinv[n * DQ + d];
  float c2 = cosv[n * DQ + d + 32], s2 = sinv[n * DQ + d + 32];
  {
    float x1 = (float)q[base + d], x2 = (float)q[base + d + 32];
    q[base + d]      = (__bf16)(x1 * c1 - x2 * s1);
    q[base + d + 32] = (__bf16)(x2 * c2 + x1 * s2);
  }
  {
    float x1 = (float)k[base + d], x2 = (float)k[base + d + 32];
    k[base + d]      = (__bf16)(x1 * c1 - x2 * s1);
    k[base + d + 32] = (__bf16)(x2 * c2 + x1 * s2);
  }
}

// =====================================================================
// Flash attention: one block = (b,h) x 128 q rows. 8 waves, each owns a
// 16-row strip: 4 S accums (16x64 scores) + 4 O accums (16x64 output).
// Q/K/V tiles staged via CDNA5 async global->LDS; online softmax with
// half-wave shuffle reductions; P relayouted through per-wave LDS.
// =====================================================================
__global__ __launch_bounds__(256) void flash_attn_kernel(
    const __bf16* __restrict__ q, const __bf16* __restrict__ k,
    const __bf16* __restrict__ v, float* __restrict__ attn_out)
{
  constexpr int LDT = 72;  // row stride (bf16): 144B, multiple of 16B
  __shared__ __bf16 sQ[128 * LDT];
  __shared__ __bf16 sK[64 * LDT];
  __shared__ __bf16 sV[64 * LDT];
  __shared__ __bf16 sP[8][16 * LDT];

  const int tid  = threadIdx.x;
  const int wave = tid >> 5, lane = tid & 31;
  const int hs = lane >> 4, ln = lane & 15;
  const int qt = blockIdx.x & 15;   // N/128 = 16 q tiles
  const int bh = blockIdx.x >> 4;   // 0..63
  const int b = bh >> 4, h = bh & 15;
  const size_t headbase = (size_t)bh * NQ * DQ;

  // async-load Q tile 128x64 (4x b128 per thread)
  {
    const __bf16* g = q + headbase + (size_t)qt * 128 * DQ;
    int row = tid >> 1, colb = (tid & 1) * 32;
#pragma unroll
    for (int c = 0; c < 4; c++) {
      int col = colb + c * 8;
      async_copy_b128(&sQ[row * LDT + col], g + row * DQ + col);
    }
  }
  wait_async0();
  __syncthreads();

  // preload this wave's Q A-fragments (d: 0..31 and 32..63)
  bf16x16 aq0, aq1;
#pragma unroll
  for (int i = 0; i < 16; i++) {
    int kk = ((i >> 3) << 4) + (hs << 3) + (i & 7);
    aq0[i] = sQ[(wave * 16 + ln) * LDT + kk];
    aq1[i] = sQ[(wave * 16 + ln) * LDT + 32 + kk];
  }

  f32x8 Oacc[4] = {};
  const f32x8 zero = {};
  float rowmax[8], rowsum[8];
#pragma unroll
  for (int r = 0; r < 8; r++) { rowmax[r] = -1e30f; rowsum[r] = 0.0f; }

  const int kvrow = tid >> 2, kvcol = (tid & 3) * 16;

  for (int j = 0; j < NQ / 64; j++) {
    __syncthreads();
    {  // async-load K,V 64x64 tiles (2x b128 per thread each)
      const __bf16* gk = k + headbase + (size_t)j * 64 * DQ;
      const __bf16* gv = v + headbase + (size_t)j * 64 * DQ;
#pragma unroll
      for (int c = 0; c < 2; c++) {
        int col = kvcol + c * 8;
        async_copy_b128(&sK[kvrow * LDT + col], gk + kvrow * DQ + col);
        async_copy_b128(&sV[kvrow * LDT + col], gv + kvrow * DQ + col);
      }
      if (j + 1 < NQ / 64) {  // prefetch next tiles into cache
        __builtin_prefetch(gk + 64 * DQ + kvrow * DQ + kvcol, 0, 1);
        __builtin_prefetch(gv + 64 * DQ + kvrow * DQ + kvcol, 0, 1);
      }
    }
    wait_async0();
    __syncthreads();

    // S = Q K^T : 16x64 strip -> 4 tiles, K-dim = 64 (2 WMMAs each)
    f32x8 Sacc[4];
#pragma unroll
    for (int ct = 0; ct < 4; ct++) {
      bf16x16 b0, b1;
#pragma unroll
      for (int i = 0; i < 16; i++) {
        int kk = ((i >> 3) << 4) + (hs << 3) + (i & 7);
        b0[i] = sK[(ct * 16 + ln) * LDT + kk];
        b1[i] = sK[(ct * 16 + ln) * LDT + 32 + kk];
      }
      f32x8 s = WMMA_BF16(aq0, b0, zero);
      s = WMMA_BF16(aq1, b1, s);
      Sacc[ct] = s;
    }

    // online softmax per accumulator row (rows split lanes 0-15/16-31,
    // so 16-lane xor reductions with masks 1..8 stay within the half)
#pragma unroll
    for (int r = 0; r < 8; r++) {
      float sl0 = Sacc[0][r] * 0.125f, sl1 = Sacc[1][r] * 0.125f;
      float sl2 = Sacc[2][r] * 0.125f, sl3 = Sacc[3][r] * 0.125f;
      float mx = fmaxf(fmaxf(sl0, sl1), fmaxf(sl2, sl3));
#pragma unroll
      for (int msk = 8; msk >= 1; msk >>= 1) mx = fmaxf(mx, __shfl_xor(mx, msk, 32));
      float mnew = fmaxf(rowmax[r], mx);
      float corr = __expf(rowmax[r] - mnew);
      float p0 = __expf(sl0 - mnew), p1 = __expf(sl1 - mnew);
      float p2 = __expf(sl2 - mnew), p3 = __expf(sl3 - mnew);
      float ps = p0 + p1 + p2 + p3;
#pragma unroll
      for (int msk = 8; msk >= 1; msk >>= 1) ps += __shfl_xor(ps, msk, 32);
      rowsum[r] = rowsum[r] * corr + ps;
      rowmax[r] = mnew;
      int R = r + hs * 8;
      __bf16* pw = &sP[wave][R * LDT];
      pw[0 * 16 + ln] = (__bf16)p0;
      pw[1 * 16 + ln] = (__bf16)p1;
      pw[2 * 16 + ln] = (__bf16)p2;
      pw[3 * 16 + ln] = (__bf16)p3;
#pragma unroll
      for (int dt = 0; dt < 4; dt++) Oacc[dt][r] *= corr;
    }

    // O += P V  (P re-read from wave-private LDS in A-fragment layout;
    // per-wave LDS ordering makes this safe without a barrier)
    bf16x16 ap0, ap1;
#pragma unroll
    for (int i = 0; i < 16; i++) {
      int kk = ((i >> 3) << 4) + (hs << 3) + (i & 7);
      ap0[i] = sP[wave][ln * LDT + kk];
      ap1[i] = sP[wave][ln * LDT + 32 + kk];
    }
#pragma unroll
    for (int dt = 0; dt < 4; dt++) {
      bf16x16 b0, b1;
#pragma unroll
      for (int i = 0; i < 16; i++) {
        int kk = ((i >> 3) << 4) + (hs << 3) + (i & 7);
        b0[i] = sV[kk * LDT + dt * 16 + ln];
        b1[i] = sV[(32 + kk) * LDT + dt * 16 + ln];
      }
      Oacc[dt] = WMMA_BF16(ap0, b0, Oacc[dt]);
      Oacc[dt] = WMMA_BF16(ap1, b1, Oacc[dt]);
    }
  }

  // finalize: divide by rowsum, store f32 attn output in [B,N,C]
#pragma unroll
  for (int r = 0; r < 8; r++) {
    float inv = 1.0f / rowsum[r];
    int n = qt * 128 + wave * 16 + r + hs * 8;
#pragma unroll
    for (int dt = 0; dt < 4; dt++) {
      int c = h * DQ + dt * 16 + ln;
      attn_out[((size_t)b * NQ + n) * CQ + c] = Oacc[dt][r] * inv;
    }
  }
}

// =====================================================================
// launch
// =====================================================================
extern "C" void kernel_launch(void* const* d_in, const int* in_sizes, int n_in,
                              void* d_out, int out_size, void* d_ws, size_t ws_size,
                              hipStream_t stream) {
  const float* x        = (const float*)d_in[0];
  const float* rope_cos = (const float*)d_in[1];
  const float* rope_sin = (const float*)d_in[2];
  const float* w_qkv    = (const float*)d_in[3];
  const float* w_proj   = (const float*)d_in[4];
  const float* b_proj   = (const float*)d_in[5];
  float* out = (float*)d_out;

  // workspace: q/k/v bf16 [B,H,N,D] (16MB each) + attn f32 [B,N,C] (32MB)
  char* ws = (char*)d_ws;
  __bf16* qb = (__bf16*)(ws);
  __bf16* kb = (__bf16*)(ws + (16ull << 20));
  __bf16* vb = (__bf16*)(ws + (32ull << 20));
  float* attn = (float*)(ws + (48ull << 20));

  const int M = BQ * NQ;  // 8192

  // 1) QKV GEMM -> scatter bf16 q/k/v in [B,H,N,D]
  dim3 g1(3 * CQ / 128, M / 128);
  gemm_xwT_kernel<<<g1, 256, 0, stream>>>(x, w_qkv, nullptr, nullptr,
                                          qb, kb, vb, M, 3 * CQ, CQ, 1);
  // 2) RoPE on q,k (in-place, pair-safe)
  rope_kernel<<<(1u << 22) / 256, 256, 0, stream>>>(qb, kb, rope_cos, rope_sin);
  // 3) flash attention -> attn f32 [B,N,C]
  flash_attn_kernel<<<BQ * HQ * (NQ / 128), 256, 0, stream>>>(qb, kb, vb, attn);
  // 4) output projection + bias -> d_out
  dim3 g4(CQ / 128, M / 128);
  gemm_xwT_kernel<<<g4, 256, 0, stream>>>(attn, w_proj, b_proj, out,
                                          nullptr, nullptr, nullptr, M, CQ, CQ, 0);
}
